// SNN_89704686944331
// MI455X (gfx1250) — matrix operand order repaired and checked by reading, not compiled, not run on hardware
//
#include <hip/hip_runtime.h>
#include <hip/hip_bf16.h>
#include <stdint.h>

#define T_TABLES 26
#define V_EMB    100000
#define D_EMB    64
#define D_DENSE  512
#define N_BATCH  4096
#define L_BAG    50
#define H_OUT    512
#define D_FEAT   (D_EMB * T_TABLES + D_EMB)   /* 1728 (logical) */
#define D_FEATP  2048                         /* padded feature pitch (K % 256 == 0) */

#define KCHUNK   256                          /* K-chunk staged per TDM DMA   */
#define ROWPITCH 528                          /* 256*2B + 16B TDM row pad     */
#define CHUNKB   (64 * ROWPITCH)              /* 33792 B per LDS buffer       */

typedef __bf16 bf16_t;
typedef __attribute__((ext_vector_type(16))) __bf16 v16bf;
typedef __attribute__((ext_vector_type(8)))  float  v8f;
typedef __attribute__((ext_vector_type(4)))  unsigned v4u;
typedef __attribute__((ext_vector_type(8)))  int      v8i_t;
typedef __attribute__((ext_vector_type(4)))  int      v4i_t;

union FragBF { v16bf v; uint4 q[2]; };

// f32 <-> bf16 (bit manipulation; no dependence on cvt ISA availability)
__device__ __forceinline__ unsigned short f2bf_raw(float f) {
    unsigned u = __builtin_bit_cast(unsigned, f);
    u = (u + 0x7FFFu + ((u >> 16) & 1u)) >> 16;
    return (unsigned short)u;
}
__device__ __forceinline__ bf16_t f2bf(float f) {
    unsigned short h = f2bf_raw(f);
    return __builtin_bit_cast(bf16_t, h);
}
__device__ __forceinline__ float bf2f(bf16_t h) {
    unsigned u = ((unsigned)__builtin_bit_cast(unsigned short, h)) << 16;
    return __builtin_bit_cast(float, u);
}

// ---------------------------------------------------------------------------
// TDM: DMA a 2D bf16 tile [64 rows x 256 cols] (row stride strideK elems)
// from global into LDS. pad_interval = 128 DWORDs = exactly one tile row
// (512B), pad_amount = 4 DWORDs (16B)  =>  LDS row pitch 528B:
//   * affine addressing: byte(n,k) = n*528 + 2*k
//   * 528/4 = 132 = 4-bank shift per row -> 16 lanes x ds_load_b128 cover
//     all 64 banks, conflict-free.
// D# layout per CDNA5 ISA.
// Wave-level op (EXEC ignored); caller syncs via s_wait_tensorcnt + barrier.
// ---------------------------------------------------------------------------
__device__ __forceinline__ void tdm_load_2d_bf16(
    unsigned lds_off, const void* gptr,
    unsigned tensorK, unsigned tensorN, unsigned strideK)
{
    unsigned long long ga = (unsigned long long)(uintptr_t)gptr;
    v4u g0;
    g0[0] = 1u;                                    // count=1, user descriptor
    g0[1] = lds_off;                               // lds_addr (bytes)
    g0[2] = (unsigned)(ga & 0xFFFFFFFFu);          // global_addr[31:0]
    g0[3] = (unsigned)((ga >> 32) & 0x01FFFFFFu)   // global_addr[56:32]
          | (2u << 30);                            // type=2 ("image")
    v8i_t g1;
    g1[0] = (1 << 16)                              // data_size: 2 bytes
          | (1 << 20)                              // pad_enable
          | (6 << 22)                              // pad_interval: 128 DWORDs
          | (3 << 25);                             // pad_amount: 4 DWORDs (16B)
    g1[1] = (int)((tensorK & 0xFFFFu) << 16);                      // dim0[15:0]
    g1[2] = (int)(((tensorK >> 16) & 0xFFFFu) | ((tensorN & 0xFFFFu) << 16));
    g1[3] = (int)(((tensorN >> 16) & 0xFFFFu) | ((unsigned)KCHUNK << 16));
    g1[4] = 64;                                    // tile_dim1=64 (tile_dim2=0)
    g1[5] = (int)strideK;                          // tensor_dim0_stride[31:0]
    g1[6] = 0;
    g1[7] = 0;
    v4i_t g2; g2[0] = 1; g2[1] = 1; g2[2] = 0; g2[3] = 0;  // dim2=1, dim3=1
    v4i_t g3; g3[0] = 0; g3[1] = (int)(1u << 16); g3[2] = 0; g3[3] = 0; // dim4=1
#if __clang_major__ >= 23
    v8i_t gz8 = {0, 0, 0, 0, 0, 0, 0, 0};
    __builtin_amdgcn_tensor_load_to_lds(g0, g1, g2, g3, gz8, 0);
#else
    __builtin_amdgcn_tensor_load_to_lds(g0, g1, g2, g3, 0);
#endif
}

// ---------------------------------------------------------------------------
// f32 -> bf16 conversion, 4 elements/thread (b128 load, b64 store)
// ---------------------------------------------------------------------------
__global__ void cvt_f32_to_bf16x4(const float* __restrict__ src,
                                  bf16_t* __restrict__ dst, int n4) {
    int i = blockIdx.x * blockDim.x + threadIdx.x;
    if (i >= n4) return;
    float4 v = ((const float4*)src)[i];
    uint2 o;
    o.x = (unsigned)f2bf_raw(v.x) | ((unsigned)f2bf_raw(v.y) << 16);
    o.y = (unsigned)f2bf_raw(v.z) | ((unsigned)f2bf_raw(v.w) << 16);
    ((uint2*)dst)[i] = o;
}

// o_w1 [512,1728] f32 -> [512,2048] bf16, zero-padded cols 1728..2047
__global__ void cvt_pad_ow1(const float* __restrict__ src,
                            bf16_t* __restrict__ dst) {
    int i = blockIdx.x * blockDim.x + threadIdx.x;   // over 512*2048/4
    if (i >= H_OUT * (D_FEATP / 4)) return;
    int n = i / (D_FEATP / 4);
    int k = (i - n * (D_FEATP / 4)) * 4;
    uint2 o = {0u, 0u};
    if (k < D_FEAT) {
        float4 v = *(const float4*)(src + (size_t)n * D_FEAT + k);
        o.x = (unsigned)f2bf_raw(v.x) | ((unsigned)f2bf_raw(v.y) << 16);
        o.y = (unsigned)f2bf_raw(v.z) | ((unsigned)f2bf_raw(v.w) << 16);
    }
    *(uint2*)(dst + (size_t)n * D_FEATP + k) = o;
}

// Zero the feature pad columns [1728..2047] so no NaN/Inf garbage reaches
// the 0-weight products of the padded output GEMM.
__global__ void zero_feat_pad(bf16_t* __restrict__ feat) {
    int i = blockIdx.x * blockDim.x + threadIdx.x;   // over 4096*320/4
    if (i >= N_BATCH * ((D_FEATP - D_FEAT) / 4)) return;
    int b = i / ((D_FEATP - D_FEAT) / 4);
    int k = D_FEAT + (i - b * ((D_FEATP - D_FEAT) / 4)) * 4;
    uint2 z = {0u, 0u};
    *(uint2*)(feat + (size_t)b * D_FEATP + k) = z;
}

// ---------------------------------------------------------------------------
// bf16 WMMA GEMM with double-buffered TDM-staged B operand.
//   C[m, n] = relu(sum_k A[m,k] * B[n,k] + bias[n])
// A: [M,K] bf16 row-major (global), B: [N,K] bf16 row-major. K % 256 == 0.
// Block = 8 waves; block tile = 128(M) x 64(N); wave tile = 16 x 64
// (4 accumulators; A fragment reused x4 from registers).
// Wave 0 issues the TDM DMA for chunk c+1 into the alternate LDS buffer,
// all waves compute chunk c from the other buffer (ds_load_b128, affine
// addresses, immediate offsets). All 4 B fragments are loaded BEFORE the
// WMMA quartet so DScnt waits stagger (<=6,<=4,<=2,0) instead of a full
// s_wait_dscnt 0 per fragment.  Then s_wait_tensorcnt + barrier.
// Grid must be exactly (M/128)*(N/64) blocks (no partial blocks).
// Fragment layouts per CDNA5 ISA 7.12.2 (wave32).
// ---------------------------------------------------------------------------
__global__ void wmma_gemm_tdm(const bf16_t* __restrict__ A,
                              const bf16_t* __restrict__ B,
                              const float* __restrict__ bias,
                              bf16_t* __restrict__ C,
                              int K, int nSupers, int ldc) {
    __shared__ __align__(64) unsigned char ldsB[2][CHUNKB];

    int nt      = blockIdx.x % nSupers;        // 64-wide N super-tile
    int mtBlock = blockIdx.x / nSupers;
    int wave    = threadIdx.x >> 5;
    int lane    = threadIdx.x & 31;
    int lm      = lane & 15;
    int hi      = lane >> 4;                   // 0: lanes 0-15, 1: lanes 16-31
    int mt      = mtBlock * 8 + wave;

    // A fragment: lane row M=lm; low half-wave K {0..7,16..23}, high {8..15,24..31}
    const bf16_t* arow = A + (size_t)(mt * 16 + lm) * K + hi * 8;
    const bf16_t* Bsup = B + (size_t)(nt * 64) * K;
    unsigned ldsBase0 = (unsigned)(uintptr_t)&ldsB[0][0];
    unsigned ldsBase1 = (unsigned)(uintptr_t)&ldsB[1][0];

    v8f acc[4];
    acc[0] = {}; acc[1] = {}; acc[2] = {}; acc[3] = {};

    int nChunks = K / KCHUNK;
    bool issuer = threadIdx.x < 32;            // wave 0 (wave-uniform branch)

    if (issuer) {                              // prologue: stage chunk 0
        tdm_load_2d_bf16(ldsBase0, Bsup, (unsigned)K, 64u, (unsigned)K);
        __builtin_amdgcn_s_wait_tensorcnt(0);
    }
    __syncthreads();

    for (int c = 0; c < nChunks; ++c) {
        int kc = c * KCHUNK;
        if (issuer && (c + 1 < nChunks)) {     // overlap DMA of c+1 with compute
            tdm_load_2d_bf16((c & 1) ? ldsBase0 : ldsBase1,
                             Bsup + kc + KCHUNK,
                             (unsigned)(K - kc - KCHUNK), 64u, (unsigned)K);
        }
        // B fragment base: col nLocal = i*16+lm, affine LDS address
        const unsigned char* buf = ldsB[c & 1] + (size_t)lm * ROWPITCH + hi * 16;

#pragma unroll
        for (int kl = 0; kl < KCHUNK; kl += 32) {
            FragBF a;
            a.q[0] = *(const uint4*)(arow + kc + kl);
            a.q[1] = *(const uint4*)(arow + kc + kl + 16);

            FragBF b[4];
#pragma unroll
            for (int i = 0; i < 4; ++i) {      // issue all 8 ds_loads first
                const unsigned char* p = buf + i * (16 * ROWPITCH) + 2 * kl;
                b[i].q[0] = *(const uint4*)p;
                b[i].q[1] = *(const uint4*)(p + 32);
            }
#pragma unroll
            for (int i = 0; i < 4; ++i) {      // then the WMMA quartet
                acc[i] = __builtin_amdgcn_wmma_f32_16x16x32_bf16(
                    false, a.v, false, b[i].v, (short)0, acc[i], false, false);
            }
        }
        if (issuer) __builtin_amdgcn_s_wait_tensorcnt(0);
        __syncthreads();                       // next buffer ready / reads done
    }

    // Epilogue: bias + relu, store bf16. C/D layout: VGPR r -> row r (+8 hi).
    int m0 = mt * 16 + hi * 8;
#pragma unroll
    for (int i = 0; i < 4; ++i) {
        int n  = nt * 64 + i * 16 + lm;
        float bn = bias[n];
        bf16_t* ccol = C + n;
#pragma unroll
        for (int r = 0; r < 8; ++r) {
            float v = acc[i][r] + bn;
            v = v > 0.0f ? v : 0.0f;
            ccol[(size_t)(m0 + r) * ldc] = f2bf(v);
        }
    }
}

// ---------------------------------------------------------------------------
// Multi-table EmbeddingBag(sum): 16 lanes x float4 cover one 256B row.
// HBM-bound phase: 26*4096*50*256B = 1.36 GB of coalesced row reads
// (~58 us floor at 23.3 TB/s). Prefetch next bag row to hide gather latency.
// ---------------------------------------------------------------------------
__global__ void emb_bag_sum(const int* __restrict__ idx,
                            const float* __restrict__ tables,
                            bf16_t* __restrict__ feat) {
    int g  = blockIdx.x * blockDim.x + threadIdx.x;
    int d4 = g & 15;
    int s  = g >> 4;                 // flattened (table, batch) sample
    if (s >= T_TABLES * N_BATCH) return;
    int t = s / N_BATCH;
    int b = s - t * N_BATCH;

    const int*   bag = idx + (size_t)s * L_BAG;              // [T,B,L] layout
    const float* tbl = tables + (size_t)t * V_EMB * D_EMB;

    float4 acc = make_float4(0.f, 0.f, 0.f, 0.f);
    for (int j = 0; j < L_BAG; ++j) {
        if (j + 1 < L_BAG)
            __builtin_prefetch(tbl + (size_t)bag[j + 1] * D_EMB + d4 * 4, 0, 0);
        float4 v = *(const float4*)(tbl + (size_t)bag[j] * D_EMB + d4 * 4);
        acc.x += v.x; acc.y += v.y; acc.z += v.z; acc.w += v.w;
    }

    uint2 o;
    o.x = (unsigned)f2bf_raw(acc.x) | ((unsigned)f2bf_raw(acc.y) << 16);
    o.y = (unsigned)f2bf_raw(acc.z) | ((unsigned)f2bf_raw(acc.w) << 16);
    *(uint2*)(feat + (size_t)b * D_FEATP + D_EMB + t * D_EMB + d4 * 4) = o;
}

// ---------------------------------------------------------------------------
// Final Linear(512,1) + relu: one wave32 per sample, shfl_xor reduction.
// ---------------------------------------------------------------------------
__global__ void final_dot_relu(const bf16_t* __restrict__ h2,
                               const float* __restrict__ w2,
                               const float* __restrict__ b2,
                               float* __restrict__ out) {
    int wv   = blockIdx.x * (blockDim.x >> 5) + (threadIdx.x >> 5);
    int lane = threadIdx.x & 31;
    if (wv >= N_BATCH) return;
    const bf16_t* row = h2 + (size_t)wv * H_OUT;
    float s = 0.f;
    for (int k = lane; k < H_OUT; k += 32)
        s += bf2f(row[k]) * w2[k];
#pragma unroll
    for (int o = 16; o > 0; o >>= 1)
        s += __shfl_xor(s, o, 32);
    if (lane == 0) {
        float v = s + b2[0];
        out[wv] = v > 0.f ? v : 0.f;
    }
}

// ---------------------------------------------------------------------------
extern "C" void kernel_launch(void* const* d_in, const int* in_sizes, int n_in,
                              void* d_out, int out_size, void* d_ws, size_t ws_size,
                              hipStream_t stream) {
    const float* dense  = (const float*)d_in[0];
    const int*   sparse = (const int*)d_in[1];
    const float* emb    = (const float*)d_in[2];
    const float* d_w1   = (const float*)d_in[3];
    const float* d_b1   = (const float*)d_in[4];
    const float* d_w2   = (const float*)d_in[5];
    const float* d_b2   = (const float*)d_in[6];
    const float* o_w1   = (const float*)d_in[7];
    const float* o_b1   = (const float*)d_in[8];
    const float* o_w2   = (const float*)d_in[9];
    const float* o_b2   = (const float*)d_in[10];
    float* out = (float*)d_out;

    // Workspace layout (bytes, 256-aligned), ~30.6 MB total
    char* ws = (char*)d_ws;
    bf16_t* w1b   = (bf16_t*)(ws + 0);          //  512*512  bf16
    bf16_t* w2b   = (bf16_t*)(ws + 524288);     //   64*512  bf16
    bf16_t* ow1b  = (bf16_t*)(ws + 589824);     // 512*2048  bf16 (zero-padded)
    bf16_t* denb  = (bf16_t*)(ws + 2686976);    // 4096*512  bf16
    bf16_t* h1b   = (bf16_t*)(ws + 6881280);    // 4096*512  bf16
    bf16_t* featb = (bf16_t*)(ws + 11075584);   // 4096*2048 bf16
    bf16_t* h2b   = (bf16_t*)(ws + 27852800);   // 4096*512  bf16

    auto cvt = [&](const float* s, bf16_t* d, int n) {
        int n4 = n / 4;
        cvt_f32_to_bf16x4<<<(n4 + 255) / 256, 256, 0, stream>>>(s, d, n4);
    };
    cvt(d_w1, w1b,  D_DENSE * D_DENSE);
    cvt(d_w2, w2b,  D_EMB   * D_DENSE);
    cvt(dense, denb, N_BATCH * D_DENSE);
    {
        int n4 = H_OUT * (D_FEATP / 4);
        cvt_pad_ow1<<<(n4 + 255) / 256, 256, 0, stream>>>(o_w1, ow1b);
    }
    {
        int n4 = N_BATCH * ((D_FEATP - D_FEAT) / 4);
        zero_feat_pad<<<(n4 + 255) / 256, 256, 0, stream>>>(featb);
    }

    // Dense GEMM1: [4096,512] x [512,512]^T -> h1   (K=512, 2 chunks)
    wmma_gemm_tdm<<<(N_BATCH / 128) * (D_DENSE / 64), 256, 0, stream>>>(
        denb, w1b, d_b1, h1b, D_DENSE, D_DENSE / 64, D_DENSE);

    // Dense GEMM2: [4096,512] x [64,512]^T -> features cols 0..63
    wmma_gemm_tdm<<<(N_BATCH / 128) * (D_EMB / 64), 256, 0, stream>>>(
        h1b, w2b, d_b2, featb, D_DENSE, D_EMB / 64, D_FEATP);

    // Embedding bags -> features cols 64..1727  (dominant, HBM-bound)
    {
        int total = T_TABLES * N_BATCH * 16;
        emb_bag_sum<<<(total + 255) / 256, 256, 0, stream>>>(sparse, emb, featb);
    }

    // Output GEMM: [4096,2048pad] x [512,2048pad]^T -> h2   (K=2048, 8 chunks)
    wmma_gemm_tdm<<<(N_BATCH / 128) * (H_OUT / 64), 256, 0, stream>>>(
        featb, ow1b, o_b1, h2b, D_FEATP, H_OUT / 64, H_OUT);

    // Final Linear(512,1)+relu
    final_dot_relu<<<(N_BATCH + 7) / 8, 256, 0, stream>>>(h2b, o_w2, o_b2, out);
}
